// PointTransformerClsClean_52527450030545
// MI455X (gfx1250) — matrix operand order repaired and checked by reading, not compiled, not run on hardware
//
#include <hip/hip_runtime.h>
#include <hip/hip_bf16.h>
#include <stdint.h>

typedef _Float16 half_t;
typedef __attribute__((ext_vector_type(16))) _Float16 v16h;
typedef __attribute__((ext_vector_type(8)))  float    v8f;

// ---------------- model dimensions (from reference setup) ----------------
#define BATCH 16
#define NALL  4096
#define M0P   2048
#define M1P   1024
#define M2P   512
#define KNN   32

// ---- CDNA5 async memory->LDS copy (ASYNCcnt-tracked, bypasses VGPRs) ----
__device__ __forceinline__ void async_load_lds_b128(void* lds_ptr, const void* gptr) {
  uint32_t lds_off = (uint32_t)(uintptr_t)lds_ptr;   // LDS address: low 32 bits of the pointer
  asm volatile("global_load_async_to_lds_b128 %0, %1, off"
               :: "v"(lds_off), "v"((uint64_t)(uintptr_t)gptr)
               : "memory");
}
__device__ __forceinline__ void wait_asynccnt0() {
  asm volatile("s_wait_asynccnt 0x0" ::: "memory");
}

// =========================================================================
// f32 -> f16 conversion (weights)
// =========================================================================
__global__ __launch_bounds__(256) void f32_to_f16_kernel(const float* __restrict__ s,
                                                         half_t* __restrict__ d, size_t n) {
  size_t i = (size_t)blockIdx.x * blockDim.x + threadIdx.x;
  if (i < n) d[i] = (half_t)s[i];
}

// =========================================================================
// emb layer 1: gather points by idx0, 3->64 linear + affine + relu, emit f16
// =========================================================================
__global__ __launch_bounds__(256) void emb1_kernel(const float* __restrict__ pts,
                                                   const int* __restrict__ idx0,
                                                   const float* __restrict__ W,   // [64,3]
                                                   const float* __restrict__ g,
                                                   const float* __restrict__ b,
                                                   half_t* __restrict__ out) {   // [B*M0,64]
  size_t i = (size_t)blockIdx.x * blockDim.x + threadIdx.x;
  size_t total = (size_t)BATCH * M0P * 64;
  if (i >= total) return;
  int d = (int)(i & 63);
  size_t row = i >> 6;
  int bb = (int)(row / M0P), m = (int)(row % M0P);
  int src = idx0[(size_t)bb * M0P + m];
  const float* p = pts + ((size_t)bb * NALL + src) * 3;
  float v = p[0] * W[d * 3 + 0] + p[1] * W[d * 3 + 1] + p[2] * W[d * 3 + 2];
  v = v * g[d] + b[d];
  out[i] = (half_t)fmaxf(v, 0.0f);
}

// =========================================================================
// Generic wave-level WMMA GEMM:  Y = epi( (A [- A2]) @ op(B) )
//   A : [R, K] f16 row-major (batched via sA)
//   B : transB==0 -> stored [N, K] row-major (weights W[d,c]); B[k][n]=W[n*ldb+k]
//       transB==1 -> stored [K, N] row-major;                 B[k][n]=B[k*ldb+n]
//   epilogue: (+bias0[n]) * g[n] + b[n], act(0 none,1 relu,2 leaky0.2),
//             + resid, store Y16 (f16, optionally transposed) and/or Y32 (f32)
//   One wave per block; each wave computes a 16 x (16*NT) tile.
// =========================================================================
struct GemmP {
  const half_t* A;  size_t sA;  int lda;
  const half_t* A2; size_t sA2; int lda2;
  const half_t* B;  size_t sB;  int ldb; int transB;
  const float* bias0; const float* gw; const float* bw; int act;
  const half_t* resid; size_t sR; int ldr;
  half_t* Y16; size_t sY16; int ldy; int yTrans;   // yTrans: Y16[n*ldy+row]
  float*  Y32; size_t sY32; int ldy32;
  int R, N, K, batches;
};

template <int NT>
__global__ __launch_bounds__(32) void gemm_wmma_kernel(GemmP p) {
  const int batch = blockIdx.z;
  const int row0 = blockIdx.y * 16;
  const int colBase = blockIdx.x * (16 * NT);
  const int lane = threadIdx.x & 31;
  const int lm = lane & 15, lh = lane >> 4;

  const half_t* Ab = p.A + (size_t)batch * p.sA;
  const half_t* A2b = p.A2 ? p.A2 + (size_t)batch * p.sA2 : nullptr;
  const half_t* Bb = p.B + (size_t)batch * p.sB;

  v8f acc[NT] = {};

  for (int k0 = 0; k0 < p.K; k0 += 32) {
    // ---- A fragment (16x32 f16), optional A - A2 ----
    v16h a;
    {
      const half_t* p0 = Ab + (size_t)(row0 + lm) * p.lda + k0 + lh * 16;
      if (k0 + 32 < p.K) __builtin_prefetch(p0 + 32, 0, 0);  // global_prefetch
#pragma unroll
      for (int j = 0; j < 8; ++j) { a[j] = p0[j]; a[8 + j] = p0[8 + j]; }
      if (A2b) {
        const half_t* q0 = A2b + (size_t)(row0 + lm) * p.lda2 + k0 + lh * 16;
#pragma unroll
        for (int j = 0; j < 8; ++j) {
          a[j]     = (half_t)((float)a[j]     - (float)q0[j]);
          a[8 + j] = (half_t)((float)a[8 + j] - (float)q0[8 + j]);
        }
      }
    }
    // ---- B fragments + WMMA ----
#pragma unroll
    for (int t = 0; t < NT; ++t) {
      int n = colBase + t * 16 + lm;
      int nc = (n < p.N) ? n : (p.N - 1);   // clamp for padded N (lin3)
      v16h bf;
      if (!p.transB) {
        const half_t* pw = Bb + (size_t)nc * p.ldb + k0 + lh * 16;
#pragma unroll
        for (int j = 0; j < 16; ++j) bf[j] = pw[j];
      } else {
        const half_t* pw = Bb + (size_t)(k0 + lh * 16) * p.ldb + nc;
#pragma unroll
        for (int j = 0; j < 16; ++j) bf[j] = pw[(size_t)j * p.ldb];
      }
      acc[t] = __builtin_amdgcn_wmma_f32_16x16x32_f16(false, a, false, bf,
                                                      (short)0, acc[t], false, false);
    }
  }

  // ---- epilogue ----
#pragma unroll
  for (int t = 0; t < NT; ++t) {
    int n = colBase + t * 16 + lm;
    if (n >= p.N) continue;
    float b0 = p.bias0 ? p.bias0[n] : 0.0f;
    float gg = p.gw ? p.gw[n] : 1.0f;
    float bb = p.bw ? p.bw[n] : 0.0f;
#pragma unroll
    for (int r = 0; r < 8; ++r) {
      int row = row0 + lh * 8 + r;
      float v = acc[t][r] + b0;
      v = v * gg + bb;
      if (p.act == 1) v = fmaxf(v, 0.0f);
      else if (p.act == 2) v = (v > 0.0f) ? v : 0.2f * v;
      if (p.resid) v += (float)p.resid[(size_t)batch * p.sR + (size_t)row * p.ldr + n];
      if (p.Y16) {
        size_t yi = p.yTrans ? ((size_t)n * p.ldy + row) : ((size_t)row * p.ldy + n);
        p.Y16[(size_t)batch * p.sY16 + yi] = (half_t)v;
      }
      if (p.Y32) p.Y32[(size_t)batch * p.sY32 + (size_t)row * p.ldy32 + n] = v;
    }
  }
}

// =========================================================================
// Fused point-conv: gather KNN rows + concat center (async global->LDS),
// MLP(2 layers, WMMA from LDS), max over K=32 neighbors.
// Block = 256 threads (8 waves), 4 centers (128 rows).
// =========================================================================
__global__ __launch_bounds__(256) void point_conv_kernel(
    const half_t* __restrict__ featIn, int Cfeat, int NinPts,   // [B,NinPts,Cfeat]
    const int* __restrict__ idx, int M,                         // [B,M,KNN]
    const half_t* __restrict__ W1, const float* __restrict__ g1,
    const float* __restrict__ b1, int Cmid,                     // W1 [Cmid, 2*Cfeat]
    const half_t* __restrict__ W2, const float* __restrict__ g2,
    const float* __restrict__ b2, int Dout,                     // W2 [Dout, Cmid]
    half_t* __restrict__ out, int ldo, size_t strideOutB) {
  extern __shared__ char smem_raw[];
  half_t* smem = (half_t*)smem_raw;
  const int Cin = 2 * Cfeat;
  const int ROWS = 128;                       // 4 centers * 32 neighbors
  const int reg1 = ROWS * ((Cin > Dout) ? Cin : Dout);
  half_t* lds_in = smem;                      // [ROWS, Cin], later reused as [ROWS, Dout]
  half_t* lds_mid = smem + reg1;              // [ROWS, Cmid]

  const int b = blockIdx.z;
  const int m0 = blockIdx.x * 4;
  const int tid = threadIdx.x;

  // ---- gather (concat [neighbor, center]) straight into LDS ----
  // All 256 lanes stay uniformly active (ROWS*Cin is a multiple of 2048),
  // so EXEC is all-ones for the async-to-LDS ops.
  for (int e = tid * 8; e < ROWS * Cin; e += blockDim.x * 8) {
    int r = e / Cin, c = e % Cin;
    int ci = r >> 5, kk = r & 31;
    size_t ibase = ((size_t)b * M + (m0 + ci)) * KNN;
    int src = (c < Cfeat) ? idx[ibase + kk] : idx[ibase];   // idx[...,0] is center
    int cc = (c < Cfeat) ? c : (c - Cfeat);
    const half_t* gp = featIn + ((size_t)b * NinPts + src) * Cfeat + cc;
    async_load_lds_b128(lds_in + (size_t)r * Cin + c, gp);  // 8 f16 per lane
  }
  wait_asynccnt0();      // our wave's async copies landed in LDS
  __syncthreads();       // all waves' copies visible

  const int wave = tid >> 5;
  const int lane = tid & 31, lm = lane & 15, lh = lane >> 4;

  // ---- layer 1: [128,Cin] @ W1^T -> relu affine -> lds_mid [128,Cmid] ----
  {
    const int mt = ROWS / 16, nt = Cmid / 16;
    for (int t = wave; t < mt * nt; t += 8) {
      int row0 = (t % mt) * 16, col0 = (t / mt) * 16;
      v8f acc = {};
      for (int k0 = 0; k0 < Cin; k0 += 32) {
        v16h a;
        const half_t* p0 = lds_in + (size_t)(row0 + lm) * Cin + k0 + lh * 16;
#pragma unroll
        for (int j = 0; j < 8; ++j) { a[j] = p0[j]; a[8 + j] = p0[8 + j]; }
        v16h bf;
        const half_t* pw = W1 + (size_t)(col0 + lm) * Cin + k0 + lh * 16;
#pragma unroll
        for (int j = 0; j < 16; ++j) bf[j] = pw[j];
        acc = __builtin_amdgcn_wmma_f32_16x16x32_f16(false, a, false, bf,
                                                     (short)0, acc, false, false);
      }
      int n = col0 + lm;
      float gg = g1[n], bb = b1[n];
#pragma unroll
      for (int r = 0; r < 8; ++r) {
        float v = fmaxf(acc[r] * gg + bb, 0.0f);
        lds_mid[(size_t)(row0 + lh * 8 + r) * Cmid + n] = (half_t)v;
      }
    }
  }
  __syncthreads();

  // ---- layer 2: [128,Cmid] @ W2^T -> relu affine -> lds_in region [128,Dout] ----
  {
    const int mt = ROWS / 16, nt = Dout / 16;
    for (int t = wave; t < mt * nt; t += 8) {
      int row0 = (t % mt) * 16, col0 = (t / mt) * 16;
      v8f acc = {};
      for (int k0 = 0; k0 < Cmid; k0 += 32) {
        v16h a;
        const half_t* p0 = lds_mid + (size_t)(row0 + lm) * Cmid + k0 + lh * 16;
#pragma unroll
        for (int j = 0; j < 8; ++j) { a[j] = p0[j]; a[8 + j] = p0[8 + j]; }
        v16h bf;
        const half_t* pw = W2 + (size_t)(col0 + lm) * Cmid + k0 + lh * 16;
#pragma unroll
        for (int j = 0; j < 16; ++j) bf[j] = pw[j];
        acc = __builtin_amdgcn_wmma_f32_16x16x32_f16(false, a, false, bf,
                                                     (short)0, acc, false, false);
      }
      int n = col0 + lm;
      float gg = g2[n], bb = b2[n];
#pragma unroll
      for (int r = 0; r < 8; ++r) {
        float v = fmaxf(acc[r] * gg + bb, 0.0f);
        lds_in[(size_t)(row0 + lh * 8 + r) * Dout + n] = (half_t)v;
      }
    }
  }
  __syncthreads();

  // ---- max over 32 neighbors per center ----
  for (int e = tid; e < 4 * Dout; e += blockDim.x) {
    int ci = e / Dout, d = e % Dout;
    float best = -3.4e38f;
#pragma unroll 4
    for (int kk = 0; kk < 32; ++kk)
      best = fmaxf(best, (float)lds_in[(size_t)(ci * 32 + kk) * Dout + d]);
    out[strideOutB * b + (size_t)(m0 + ci) * ldo + d] = (half_t)best;
  }
}

// =========================================================================
// Attention pieces (f32 for numerics)
// =========================================================================
__global__ __launch_bounds__(256) void softmax_row_kernel(float* __restrict__ e) {
  // one block per row of length 512
  __shared__ float red[256];
  float* row = e + (size_t)blockIdx.x * 512;
  int t = threadIdx.x;
  float v0 = row[t], v1 = row[t + 256];
  red[t] = fmaxf(v0, v1);
  __syncthreads();
  for (int s = 128; s > 0; s >>= 1) { if (t < s) red[t] = fmaxf(red[t], red[t + s]); __syncthreads(); }
  float mx = red[0];
  __syncthreads();
  v0 = __expf(v0 - mx); v1 = __expf(v1 - mx);
  red[t] = v0 + v1;
  __syncthreads();
  for (int s = 128; s > 0; s >>= 1) { if (t < s) red[t] += red[t + s]; __syncthreads(); }
  float inv = 1.0f / red[0];
  row[t] = v0 * inv;
  row[t + 256] = v1 * inv;
}

__global__ __launch_bounds__(256) void colsum_kernel(const float* __restrict__ a,
                                                     float* __restrict__ s) {
  int n = blockIdx.x * blockDim.x + threadIdx.x;  // column (512)
  int b = blockIdx.y;
  if (n >= 512) return;
  const float* p = a + ((size_t)b * 512) * 512 + n;
  float acc = 0.0f;
  for (int m = 0; m < 512; ++m) acc += p[(size_t)m * 512];
  s[(size_t)b * 512 + n] = acc;
}

__global__ __launch_bounds__(256) void norm_transpose_kernel(const float* __restrict__ a,
                                                             const float* __restrict__ s,
                                                             half_t* __restrict__ aT) {
  size_t i = (size_t)blockIdx.x * blockDim.x + threadIdx.x;
  size_t total = (size_t)BATCH * 512 * 512;
  if (i >= total) return;
  int n = (int)(i & 511);
  size_t r = i >> 9;
  int m = (int)(r & 511);
  int b = (int)(r >> 9);
  float v = a[i] / (1e-9f + s[(size_t)b * 512 + n]);
  aT[((size_t)b * 512 + n) * 512 + m] = (half_t)v;   // aT[b][n][m] = a[b][m][n]
}

__global__ __launch_bounds__(256) void pool_max_kernel(const half_t* __restrict__ fused,
                                                       float* __restrict__ p32,
                                                       half_t* __restrict__ p16) {
  int d = blockIdx.x * blockDim.x + threadIdx.x;  // 1024
  int b = blockIdx.y;
  if (d >= 1024) return;
  const half_t* p = fused + ((size_t)b * 512) * 1024 + d;
  float best = -3.4e38f;
  for (int r = 0; r < 512; ++r) best = fmaxf(best, (float)p[(size_t)r * 1024]);
  p32[(size_t)b * 1024 + d] = best;
  p16[(size_t)b * 1024 + d] = (half_t)best;
}

// =========================================================================
// host side
// =========================================================================
static void run_gemm(const GemmP& p, hipStream_t st) {
  int tiles = (p.N + 15) / 16;
  int NT = (tiles % 4 == 0) ? 4 : (tiles % 3 == 0) ? 3 : (tiles % 2 == 0) ? 2 : 1;
  dim3 grid(tiles / NT, p.R / 16, p.batches);
  dim3 blk(32, 1, 1);
  switch (NT) {
    case 4: hipLaunchKernelGGL((gemm_wmma_kernel<4>), grid, blk, 0, st, p); break;
    case 3: hipLaunchKernelGGL((gemm_wmma_kernel<3>), grid, blk, 0, st, p); break;
    case 2: hipLaunchKernelGGL((gemm_wmma_kernel<2>), grid, blk, 0, st, p); break;
    default: hipLaunchKernelGGL((gemm_wmma_kernel<1>), grid, blk, 0, st, p); break;
  }
}

extern "C" void kernel_launch(void* const* d_in, const int* in_sizes, int n_in,
                              void* d_out, int out_size, void* d_ws, size_t ws_size,
                              hipStream_t stream) {
  (void)in_sizes; (void)n_in; (void)out_size; (void)ws_size;
  // ---- input leaf mapping (jax sorted-key pytree flatten order) ----
  // 0 all_points, 1 idx0, 2 idx1, 3 idx2, then params (alphabetical):
  // conv1: 4 W0, 5 g0, 6 b0, 7 W1, 8 g1, 9 b1
  // conv2: 10 W0, 11 g0, 12 b0, 13 W1, 14 g1, 15 b1
  // emb:   16 W0, 17 g0, 18 b0, 19 W1, 20 g1, 21 b1
  // fuse:  22 W, 23 g, 24 b
  // l1: 25 W, 26 g, 27 b        l2: 28 W, 29 g, 30 b
  // lin1: 31 W, 32 g, 33 b      lin2: 34 W, 35 bb, 36 g, 37 b    lin3: 38 W, 39 b
  // sa[i] (i=0..3), base=40+7i: +0 Wqk, +1 Wt, +2 Wv, +3 b, +4 bt, +5 bv, +6 g
  auto F = [&](int i) { return (const float*)d_in[i]; };
  const float* pts = (const float*)d_in[0];
  const int* idx0 = (const int*)d_in[1];
  const int* idx1 = (const int*)d_in[2];
  const int* idx2 = (const int*)d_in[3];
  float* out = (float*)d_out;

  // ---- workspace bump allocator ----
  char* ws = (char*)d_ws;
  size_t off = 0;
  auto alloc = [&](size_t bytes) -> char* {
    off = (off + 255) & ~(size_t)255;
    char* p = ws + off;
    off += bytes;
    return p;
  };
  auto aH = [&](size_t n) { return (half_t*)alloc(n * sizeof(half_t)); };
  auto aF = [&](size_t n) { return (float*)alloc(n * sizeof(float)); };

  // f16 weight arena
  half_t* w_emb2 = aH(64 * 64);
  half_t* w_c1a = aH(128 * 128);
  half_t* w_c1b = aH(128 * 128);
  half_t* w_c2a = aH(128 * 256);
  half_t* w_c2b = aH(256 * 128);
  half_t* w_l1 = aH(256 * 256);
  half_t* w_l2 = aH(256 * 256);
  half_t* w_fuse = aH(1024 * 1280);
  half_t* w_lin1 = aH(512 * 1024);
  half_t* w_lin2 = aH(256 * 512);
  half_t* w_lin3 = aH(40 * 256);
  half_t* w_qk[4]; half_t* w_t[4]; half_t* w_v[4];
  for (int i = 0; i < 4; ++i) { w_qk[i] = aH(64 * 256); w_t[i] = aH(256 * 256); w_v[i] = aH(256 * 256); }

  // activations
  half_t* feat0 = aH((size_t)BATCH * M0P * 64);
  half_t* feat = aH((size_t)BATCH * M0P * 64);
  half_t* f1 = aH((size_t)BATCH * M1P * 128);
  half_t* cat = aH((size_t)BATCH * M2P * 1280);   // [sa0|sa1|sa2|sa3|f2]
  half_t* tmpA = aH((size_t)BATCH * M2P * 256);
  half_t* xcur = aH((size_t)BATCH * M2P * 256);
  half_t* q16 = aH((size_t)BATCH * M2P * 64);
  half_t* vT16 = aH((size_t)BATCH * 256 * 512);   // v transposed: [c, m]
  half_t* xr16 = aH((size_t)BATCH * M2P * 256);
  float* ebuf = aF((size_t)BATCH * 512 * 512);
  float* csum = aF((size_t)BATCH * 512);
  half_t* aT16 = aH((size_t)BATCH * 512 * 512);
  half_t* fused16 = aH((size_t)BATCH * 512 * 1024);
  float* pooled32 = aF((size_t)BATCH * 1024);
  half_t* pooled16 = aH((size_t)BATCH * 1024);
  half_t* lin1o = aH((size_t)BATCH * 512);
  half_t* lin2o = aH((size_t)BATCH * 256);

  // ---- weight conversion ----
  auto cvt = [&](const float* s, half_t* d, size_t n) {
    f32_to_f16_kernel<<<dim3((unsigned)((n + 255) / 256)), dim3(256), 0, stream>>>(s, d, n);
  };
  cvt(F(19), w_emb2, 64 * 64);
  cvt(F(4), w_c1a, 128 * 128);  cvt(F(7), w_c1b, 128 * 128);
  cvt(F(10), w_c2a, 128 * 256); cvt(F(13), w_c2b, 256 * 128);
  cvt(F(25), w_l1, 256 * 256);  cvt(F(28), w_l2, 256 * 256);
  cvt(F(22), w_fuse, 1024 * 1280);
  cvt(F(31), w_lin1, 512 * 1024);
  cvt(F(34), w_lin2, 256 * 512);
  cvt(F(38), w_lin3, 40 * 256);
  for (int i = 0; i < 4; ++i) {
    int base = 40 + 7 * i;
    cvt(F(base + 0), w_qk[i], 64 * 256);
    cvt(F(base + 1), w_t[i], 256 * 256);
    cvt(F(base + 2), w_v[i], 256 * 256);
  }

  // ---- embedding ----
  {
    size_t tot = (size_t)BATCH * M0P * 64;
    emb1_kernel<<<dim3((unsigned)((tot + 255) / 256)), dim3(256), 0, stream>>>(
        pts, idx0, F(16), F(17), F(18), feat0);
    GemmP p{}; p.A = feat0; p.lda = 64; p.B = w_emb2; p.ldb = 64;
    p.gw = F(20); p.bw = F(21); p.act = 1;
    p.Y16 = feat; p.ldy = 64;
    p.R = BATCH * M0P; p.N = 64; p.K = 64; p.batches = 1;
    run_gemm(p, stream);
  }

  // ---- point convs ----
  {
    size_t sm1 = ((size_t)128 * 128 + 128 * 128) * sizeof(half_t);   // 64 KB
    point_conv_kernel<<<dim3(M1P / 4, 1, BATCH), dim3(256), sm1, stream>>>(
        feat, 64, M0P, idx1, M1P,
        w_c1a, F(5), F(6), 128,
        w_c1b, F(8), F(9), 128,
        f1, 128, (size_t)M1P * 128);
    size_t sm2 = ((size_t)128 * 256 + 128 * 128) * sizeof(half_t);   // 96 KB
    point_conv_kernel<<<dim3(M2P / 4, 1, BATCH), dim3(256), sm2, stream>>>(
        f1, 128, M1P, idx2, M2P,
        w_c2a, F(11), F(12), 128,
        w_c2b, F(14), F(15), 256,
        cat + 1024, 1280, (size_t)M2P * 1280);   // f2 -> concat cols [1024:1280]
  }

  // ---- l1, l2 ----
  {
    GemmP p{}; p.A = cat + 1024; p.sA = (size_t)M2P * 1280; p.lda = 1280;
    p.B = w_l1; p.ldb = 256; p.gw = F(26); p.bw = F(27); p.act = 1;
    p.Y16 = tmpA; p.sY16 = (size_t)M2P * 256; p.ldy = 256;
    p.R = M2P; p.N = 256; p.K = 256; p.batches = BATCH;
    run_gemm(p, stream);
    GemmP p2{}; p2.A = tmpA; p2.sA = (size_t)M2P * 256; p2.lda = 256;
    p2.B = w_l2; p2.ldb = 256; p2.gw = F(29); p2.bw = F(30); p2.act = 1;
    p2.Y16 = xcur; p2.sY16 = (size_t)M2P * 256; p2.ldy = 256;
    p2.R = M2P; p2.N = 256; p2.K = 256; p2.batches = BATCH;
    run_gemm(p2, stream);
  }

  // ---- 4 offset-attention layers ----
  for (int i = 0; i < 4; ++i) {
    int base = 40 + 7 * i;
    const half_t* xp = (i == 0) ? xcur : (cat + (size_t)(i - 1) * 256);
    int ldx = (i == 0) ? 256 : 1280;
    size_t sx = (size_t)M2P * ldx;

    // q = x @ Wqk^T
    GemmP pq{}; pq.A = xp; pq.sA = sx; pq.lda = ldx;
    pq.B = w_qk[i]; pq.ldb = 256;
    pq.Y16 = q16; pq.sY16 = (size_t)M2P * 64; pq.ldy = 64;
    pq.R = M2P; pq.N = 64; pq.K = 256; pq.batches = BATCH;
    run_gemm(pq, stream);

    // e[m,n] = q[m,:] . q[n,:]  (B = q in N-major layout)
    GemmP pe{}; pe.A = q16; pe.sA = (size_t)M2P * 64; pe.lda = 64;
    pe.B = q16; pe.sB = (size_t)M2P * 64; pe.ldb = 64;
    pe.Y32 = ebuf; pe.sY32 = (size_t)M2P * 512; pe.ldy32 = 512;
    pe.R = M2P; pe.N = 512; pe.K = 64; pe.batches = BATCH;
    run_gemm(pe, stream);

    // softmax over n, column renorm over m, emit transposed f16
    softmax_row_kernel<<<dim3(BATCH * 512), dim3(256), 0, stream>>>(ebuf);
    colsum_kernel<<<dim3(2, BATCH), dim3(256), 0, stream>>>(ebuf, csum);
    {
      size_t tot = (size_t)BATCH * 512 * 512;
      norm_transpose_kernel<<<dim3((unsigned)((tot + 255) / 256)), dim3(256), 0, stream>>>(
          ebuf, csum, aT16);
    }

    // v = x @ Wv^T + bv, stored transposed: vT[c, m]
    GemmP pv{}; pv.A = xp; pv.sA = sx; pv.lda = ldx;
    pv.B = w_v[i]; pv.ldb = 256; pv.bias0 = F(base + 5);
    pv.Y16 = vT16; pv.sY16 = (size_t)256 * 512; pv.ldy = 512; pv.yTrans = 1;
    pv.R = M2P; pv.N = 256; pv.K = 256; pv.batches = BATCH;
    run_gemm(pv, stream);

    // x_r[j,c] = sum_m aT[j,m] * v[m,c] = sum_m aT[j,m] * vT[c,m]  (contiguous B)
    GemmP pr{}; pr.A = aT16; pr.sA = (size_t)512 * 512; pr.lda = 512;
    pr.B = vT16; pr.sB = (size_t)256 * 512; pr.ldb = 512;
    pr.Y16 = xr16; pr.sY16 = (size_t)M2P * 256; pr.ldy = 256;
    pr.R = M2P; pr.N = 256; pr.K = 512; pr.batches = BATCH;
    run_gemm(pr, stream);

    // x_out = x + relu(((x - x_r) @ Wt^T + bt) * g + b)  -> cat cols [i*256 : i*256+256]
    GemmP pt{}; pt.A = xp; pt.sA = sx; pt.lda = ldx;
    pt.A2 = xr16; pt.sA2 = (size_t)M2P * 256; pt.lda2 = 256;
    pt.B = w_t[i]; pt.ldb = 256;
    pt.bias0 = F(base + 4); pt.gw = F(base + 6); pt.bw = F(base + 3); pt.act = 1;
    pt.resid = xp; pt.sR = sx; pt.ldr = ldx;
    pt.Y16 = cat + (size_t)i * 256; pt.sY16 = (size_t)M2P * 1280; pt.ldy = 1280;
    pt.R = M2P; pt.N = 256; pt.K = 256; pt.batches = BATCH;
    run_gemm(pt, stream);
  }

  // ---- fuse: leaky_relu(cat @ Wf^T * g + b) ----
  {
    GemmP p{}; p.A = cat; p.lda = 1280;
    p.B = w_fuse; p.ldb = 1280; p.gw = F(23); p.bw = F(24); p.act = 2;
    p.Y16 = fused16; p.ldy = 1024;
    p.R = BATCH * M2P; p.N = 1024; p.K = 1280; p.batches = 1;
    run_gemm(p, stream);
  }
  pool_max_kernel<<<dim3(4, BATCH), dim3(256), 0, stream>>>(fused16, pooled32, pooled16);

  // ---- head ----
  {
    GemmP p{}; p.A = pooled16; p.lda = 1024;
    p.B = w_lin1; p.ldb = 1024; p.gw = F(32); p.bw = F(33); p.act = 1;
    p.Y16 = lin1o; p.ldy = 512;
    p.R = BATCH; p.N = 512; p.K = 1024; p.batches = 1;
    run_gemm(p, stream);
    GemmP p2{}; p2.A = lin1o; p2.lda = 512;
    p2.B = w_lin2; p2.ldb = 512; p2.bias0 = F(35); p2.gw = F(36); p2.bw = F(37); p2.act = 1;
    p2.Y16 = lin2o; p2.ldy = 256;
    p2.R = BATCH; p2.N = 256; p2.K = 512; p2.batches = 1;
    run_gemm(p2, stream);
    GemmP p3{}; p3.A = lin2o; p3.lda = 256;
    p3.B = w_lin3; p3.ldb = 256; p3.bias0 = F(39); p3.act = 0;
    p3.Y32 = out; p3.ldy32 = 40;
    p3.R = BATCH; p3.N = 40; p3.K = 256; p3.batches = 1;   // tiles=3 -> NT=3, col-guarded
    run_gemm(p3, stream);
  }
}
